// LSTM_72971494359201
// MI455X (gfx1250) — matrix operand order repaired and checked by reading, not compiled, not run on hardware
//
#include <hip/hip_runtime.h>

// Recurrent cell: per step g = [h | x_t] @ W + b  (M=128, K=513, N=512),
// sg = sigmoid(g); c' = sg*(c + tanh(g)); h' = sg*tanh(c'); out[:,t,:] = h'.
// Latency-bound scan -> persistent grid, W resident in LDS (bf16, WMMA-swizzled),
// c in registers, h ping-pong in L2-backed workspace, atomic grid barrier per step.

#define B_SZ    128
#define T_LEN   2048
#define H_DIM   512
#define NWG     16          // workgroups; each owns NT columns of N
#define NT      32
#define THREADS 256         // 8 waves
#define KCH     (H_DIM / 32)  // 16 K-chunks of 32

typedef __attribute__((ext_vector_type(16))) __bf16 v16bf;
typedef __attribute__((ext_vector_type(8)))  float  v8f;
typedef unsigned int u32x4 __attribute__((ext_vector_type(4)));

union Frag { v16bf v; u32x4 q[2]; };

__device__ __forceinline__ unsigned short f2bf(float f) {  // RNE f32 -> bf16 bits
    unsigned u = __builtin_bit_cast(unsigned, f);
    u += 0x7FFFu + ((u >> 16) & 1u);
    return (unsigned short)(u >> 16);
}
__device__ __forceinline__ float sigmoidf_(float x) { return 1.f / (1.f + __expf(-x)); }
__device__ __forceinline__ float tanhf_(float x)    { return 2.f / (1.f + __expf(-2.f * x)) - 1.f; }

// h buffers are stored in WMMA A-fragment order:
// flat bf16 index = (((mt*KCH + kc)*2 + qi)*32 + lane)*8 + e
//   row = mt*16 + (lane&15)
//   col = kc*32 + qi*16 + (lane>>4)*8 + e          (A 16x32 layout, ISA 7.12.2)

__global__ void lstm_init_kernel(const float* __restrict__ h0,
                                 unsigned short* __restrict__ hb0,
                                 unsigned int* __restrict__ bar) {
    int i = blockIdx.x * blockDim.x + threadIdx.x;
    if (i < B_SZ * H_DIM) {
        int e = i & 7, lane = (i >> 3) & 31, qi = (i >> 8) & 1;
        int kc = (i >> 9) & 15, mt = (i >> 13) & 7;
        int row = mt * 16 + (lane & 15);
        int col = kc * 32 + qi * 16 + (lane >> 4) * 8 + e;
        hb0[i] = f2bf(h0[row * H_DIM + col]);
    }
    if (i == 0) *bar = 0u;   // reset grid barrier every call (graph-replay safe)
}

__global__ void __launch_bounds__(THREADS, 1)
lstm_persistent_kernel(const float* __restrict__ x,     // [B, T]
                       const float* __restrict__ W,     // [H+1, H]
                       const float* __restrict__ bias,  // [H]
                       const float* __restrict__ c0,    // [B, H]
                       float* __restrict__ out,         // [B, T, H]
                       unsigned short* __restrict__ hb0,
                       unsigned short* __restrict__ hb1,
                       unsigned int* __restrict__ bar) {
    // W slice for this WG, bf16, pre-swizzled into B-fragment order:
    // frag idx = ((kc*2 + nt)*2 + qi), 32 lanes x 16B each -> ds_load_b128 per read
    __shared__ u32x4 ldsWq[KCH * 2 * 2 * 32];   // 32 KB

    const int wg   = blockIdx.x;        // owns columns [wg*32, wg*32+32)
    const int wave = threadIdx.x >> 5;  // owns rows    [wave*16, wave*16+16)
    const int lane = threadIdx.x & 31;
    const int half = lane >> 4;
    const int l15  = lane & 15;
    const int n0   = wg * NT;
    const int m0   = wave * 16;

    // ---- stage W slice: B 32x16 bf16 layout: lanes0-15 K0..7,16..23, lanes16-31 K8..15,24..31
    for (int idx = threadIdx.x; idx < KCH * 2 * 2 * 32; idx += THREADS) {
        int ln = idx & 31;
        int qi = (idx >> 5) & 1;
        int nt = (idx >> 6) & 1;
        int kc = idx >> 7;
        int n  = n0 + nt * 16 + (ln & 15);
        int kb = kc * 32 + qi * 16 + (ln >> 4) * 8;
        union { unsigned short s[8]; u32x4 q; } tmp;
        #pragma unroll
        for (int e = 0; e < 8; ++e) tmp.s[e] = f2bf(W[(kb + e) * H_DIM + n]);
        ldsWq[idx] = tmp.q;
    }

    // per-lane constants: bias and the x-column (row H of W) for both N-subtiles
    const float bias0 = bias[n0 + l15];
    const float bias1 = bias[n0 + 16 + l15];
    const float wl0   = W[H_DIM * H_DIM + n0 + l15];
    const float wl1   = W[H_DIM * H_DIM + n0 + 16 + l15];

    // c state in C-accumulator layout (VGPR v -> row m0 + v + 8*half), lives in regs
    v8f cacc0, cacc1;
    #pragma unroll
    for (int v = 0; v < 8; ++v) {
        int row = m0 + v + 8 * half;
        cacc0[v] = c0[row * H_DIM + n0 + l15];
        cacc1[v] = c0[row * H_DIM + n0 + 16 + l15];
    }
    __syncthreads();

    const int lhb = (l15 >> 3) & 1;  // lane's byte-half within a fragment row
    const int e7  = l15 & 7;

    for (int t = 0; t < T_LEN; ++t) {
        const u32x4* __restrict__ hq = (const u32x4*)((t & 1) ? hb1 : hb0);
        unsigned short* __restrict__ hdst = (t & 1) ? hb0 : hb1;

        v8f acc0 = {}; v8f acc1 = {};
        #pragma unroll
        for (int kc = 0; kc < KCH; ++kc) {
            Frag a, b0, b1;
            // A: coalesced b128 loads from pre-swizzled h buffer
            a.q[0] = hq[((wave * KCH + kc) * 2 + 0) * 32 + lane];
            a.q[1] = hq[((wave * KCH + kc) * 2 + 1) * 32 + lane];
            // B: ds_load_b128 from resident LDS weights
            b0.q[0] = ldsWq[((kc * 2 + 0) * 2 + 0) * 32 + lane];
            b0.q[1] = ldsWq[((kc * 2 + 0) * 2 + 1) * 32 + lane];
            b1.q[0] = ldsWq[((kc * 2 + 1) * 2 + 0) * 32 + lane];
            b1.q[1] = ldsWq[((kc * 2 + 1) * 2 + 1) * 32 + lane];
            acc0 = __builtin_amdgcn_wmma_f32_16x16x32_bf16(
                       false, a.v, false, b0.v, (short)0, acc0, false, false);
            acc1 = __builtin_amdgcn_wmma_f32_16x16x32_bf16(
                       false, a.v, false, b1.v, (short)0, acc1, false, false);
        }

        // epilogue: fold x_t column (rank-1), cell nonlinearity, writebacks
        #pragma unroll
        for (int v = 0; v < 8; ++v) {
            const int row = m0 + v + 8 * half;
            const float xr = x[row * T_LEN + t];

            float g0  = acc0[v] + bias0 + xr * wl0;
            float sg0 = sigmoidf_(g0);
            float cn0 = sg0 * (cacc0[v] + tanhf_(g0));
            cacc0[v]  = cn0;
            float hn0 = sg0 * tanhf_(cn0);

            float g1  = acc1[v] + bias1 + xr * wl1;
            float sg1 = sigmoidf_(g1);
            float cn1 = sg1 * (cacc1[v] + tanhf_(g1));
            cacc1[v]  = cn1;
            float hn1 = sg1 * tanhf_(cn1);

            // streamed output (never re-read) -> non-temporal
            size_t ob = ((size_t)row * T_LEN + t) * H_DIM + n0;
            __builtin_nontemporal_store(hn0, out + ob + l15);
            __builtin_nontemporal_store(hn1, out + ob + 16 + l15);

            // h' -> ping-pong buffer in A-fragment-swizzled bf16 (mt=wave, kc=wg)
            const int slot = v + 8 * half + 16 * lhb;
            hdst[((((wave * KCH + wg) * 2 + 0) * 32 + slot) * 8) + e7] = f2bf(hn0);
            hdst[((((wave * KCH + wg) * 2 + 1) * 32 + slot) * 8) + e7] = f2bf(hn1);
        }

        // device-wide step barrier (double-buffered h makes it race-free)
        __threadfence();
        __syncthreads();
        if (threadIdx.x == 0) {
            __hip_atomic_fetch_add(bar, 1u, __ATOMIC_RELEASE, __HIP_MEMORY_SCOPE_AGENT);
            const unsigned target = (unsigned)(t + 1) * NWG;
            while (__hip_atomic_load(bar, __ATOMIC_ACQUIRE, __HIP_MEMORY_SCOPE_AGENT) < target)
                __builtin_amdgcn_s_sleep(1);
        }
        __syncthreads();
        __threadfence();
    }
}

extern "C" void kernel_launch(void* const* d_in, const int* in_sizes, int n_in,
                              void* d_out, int out_size, void* d_ws, size_t ws_size,
                              hipStream_t stream) {
    (void)in_sizes; (void)n_in; (void)out_size; (void)ws_size;
    const float* x  = (const float*)d_in[0];   // [B, T]
    const float* W  = (const float*)d_in[1];   // [H+1, H]
    const float* b  = (const float*)d_in[2];   // [H]
    const float* h0 = (const float*)d_in[3];   // [B, H]
    const float* c0 = (const float*)d_in[4];   // [B, H]
    float* out = (float*)d_out;                // [B, T, H]

    unsigned short* hb0 = (unsigned short*)d_ws;                       // 128 KB
    unsigned short* hb1 = hb0 + (size_t)B_SZ * H_DIM;                  // 128 KB
    unsigned int*   bar = (unsigned int*)((char*)d_ws +
                          2 * (size_t)B_SZ * H_DIM * sizeof(unsigned short));

    lstm_init_kernel<<<(B_SZ * H_DIM + 255) / 256, 256, 0, stream>>>(h0, hb0, bar);
    lstm_persistent_kernel<<<NWG, THREADS, 0, stream>>>(x, W, b, c0, out,
                                                        hb0, hb1, bar);
}
